// Dual_Attention_63840393888341
// MI455X (gfx1250) — matrix operand order repaired and checked by reading, not compiled, not run on hardware
//
#include <hip/hip_runtime.h>
#include <hip/hip_bf16.h>

// MI455X / gfx1250, wave32. All matmuls via v_wmma_f32_16x16x32_f16.
// Pipeline: (1) fused QKV GEMM fp32->f16, (2) flash attention (online
// softmax, async double-buffered K/V staging via global_load_async_to_lds),
// (3) projection GEMM + bias -> fp32.

typedef __attribute__((ext_vector_type(16))) _Float16 v16h;
typedef __attribute__((ext_vector_type(8)))  _Float16 v8h;
typedef __attribute__((ext_vector_type(4)))  _Float16 v4h;
typedef __attribute__((ext_vector_type(8)))  float    v8f;

// Build a 16-half A/B fragment from two contiguous 16-byte chunks.
static __device__ __forceinline__ v16h ldA(const _Float16* lo, const _Float16* hi) {
    v8h a = *(const v8h*)lo;
    v8h b = *(const v8h*)hi;
    v16h r;
#pragma unroll
    for (int i = 0; i < 8; ++i) { r[i] = a[i]; r[i + 8] = b[i]; }
    return r;
}

static __device__ __forceinline__ v8f wmma16(v16h a, v16h b, v8f c) {
    return __builtin_amdgcn_wmma_f32_16x16x32_f16(
        /*neg_a=*/false, a, /*neg_b=*/false, b,
        /*c_mod=*/(short)0, c, /*reuse_a=*/false, /*reuse_b=*/false);
}

// Async global->LDS copy, 16 bytes per lane. Tracked by ASYNCcnt.
// (ISA 15.18.3 GLOBAL_LOAD_ASYNC_TO_LDS_B128; generic LDS addr low 32 bits
// are the LDS byte offset per the aperture rules in CDNA5 ISA.)
static __device__ __forceinline__ void async_ld_b128(unsigned lds_off,
                                                     const _Float16* gptr) {
    asm volatile("global_load_async_to_lds_b128 %0, %1, off"
                 :: "v"(lds_off), "v"((unsigned long long)(__UINTPTR_TYPE__)gptr)
                 : "memory");
}
static __device__ __forceinline__ void wait_async0() {
    asm volatile("s_wait_asynccnt 0x0" ::: "memory");
}

// ---------------------------------------------------------------------------
// Kernel 1: QKV GEMM.  X[16384 x 512] (fp32, both streams) @ Wqkv[512 x 1536]
// Scatter epilogue: Q (x 1/8, f16 [s][b][h][n][d]), K (f16 [s][b][h][n][d]),
// V transposed (f16 [s][b][h][d][n]).
// ---------------------------------------------------------------------------
__global__ __launch_bounds__(256) void qkv_gemm_kernel(
    const float* __restrict__ x1, const float* __restrict__ x2,
    const float* __restrict__ w,
    _Float16* __restrict__ Qb, _Float16* __restrict__ Kb,
    _Float16* __restrict__ Vt)
{
    __shared__ __align__(16) _Float16 As[128][32];
    __shared__ __align__(16) _Float16 Bs[64][32];   // [n][k] (transposed)

    const int tid = threadIdx.x;
    const int wid = tid >> 5, lane = tid & 31;
    const int lanelo = lane & 15, halfsel = lane >> 4;
    const int wm = wid & 3, wn = wid >> 2;
    const int mb = blockIdx.x * 128;
    const int nb = blockIdx.y * 64;
    const float* __restrict__ X = (mb < 8192) ? x1 : x2;
    const int xr0 = mb & 8191;

    v8f acc[2][2] = {};

    for (int kb = 0; kb < 512; kb += 32) {
#pragma unroll
        for (int p = 0; p < 4; ++p) {
            int r = (tid >> 3) + p * 32;
            int c = (tid & 7) * 4;
            float4 f = *(const float4*)(X + (size_t)(xr0 + r) * 512 + kb + c);
            v4h hv;
            hv[0] = (_Float16)f.x; hv[1] = (_Float16)f.y;
            hv[2] = (_Float16)f.z; hv[3] = (_Float16)f.w;
            *(v4h*)&As[r][c] = hv;
        }
#pragma unroll
        for (int p = 0; p < 8; ++p) {
            int n = tid & 63;
            int k = (tid >> 6) + p * 4;
            Bs[n][k] = (_Float16)w[(size_t)(kb + k) * 1536 + nb + n];
        }
        __syncthreads();

        v16h a[2], bf[2];
#pragma unroll
        for (int i = 0; i < 2; ++i) {
            int row = wm * 32 + i * 16 + lanelo;
            a[i] = ldA(&As[row][halfsel * 8], &As[row][16 + halfsel * 8]);
        }
#pragma unroll
        for (int j = 0; j < 2; ++j) {
            int col = wn * 32 + j * 16 + lanelo;
            bf[j] = ldA(&Bs[col][halfsel * 8], &Bs[col][16 + halfsel * 8]);
        }
#pragma unroll
        for (int i = 0; i < 2; ++i)
#pragma unroll
            for (int j = 0; j < 2; ++j)
                acc[i][j] = wmma16(a[i], bf[j], acc[i][j]);
        __syncthreads();
    }

    // Scatter epilogue; which/h/s/b derived from scalar tile bases so the
    // classification branches are wave-uniform (readfirstlane -> s_cbranch).
    const int s  = mb >> 13;
    const int bb = (mb >> 12) & 1;
    const int nrow0 = (mb & 4095) + wm * 32;
#pragma unroll
    for (int j = 0; j < 2; ++j) {
        const int ncol  = nb + wn * 32 + j * 16;        // tile column base
        const int which = __builtin_amdgcn_readfirstlane(ncol >> 9);
        const int h     = (ncol >> 6) & 7;
        const int d     = (ncol & 63) + lanelo;
        const size_t head = (size_t)((s * 2 + bb) * 8 + h);
#pragma unroll
        for (int i = 0; i < 2; ++i) {
            const int nn0 = nrow0 + i * 16 + halfsel * 8;   // 8 consecutive rows
            if (which == 2) {
                v8h pk;
#pragma unroll
                for (int r = 0; r < 8; ++r) pk[r] = (_Float16)acc[i][j][r];
                *(v8h*)(Vt + (head * 64 + d) * 4096 + nn0) = pk;
            } else if (which == 0) {
                _Float16* dst = Qb + (head * 4096 + nn0) * 64 + d;
#pragma unroll
                for (int r = 0; r < 8; ++r)
                    dst[(size_t)r * 64] = (_Float16)(acc[i][j][r] * 0.125f);
            } else {
                _Float16* dst = Kb + (head * 4096 + nn0) * 64 + d;
#pragma unroll
                for (int r = 0; r < 8; ++r)
                    dst[(size_t)r * 64] = (_Float16)acc[i][j][r];
            }
        }
    }
}

// ---------------------------------------------------------------------------
// Kernel 2: flash attention. Block = (stream,b,h) x 128-query tile; 8 waves x
// 16 queries. Keys ALWAYS from stream 0 (k1); values from query's stream.
// K/V tiles double-buffered in LDS, filled by async global->LDS loads that
// overlap the WMMA + softmax work on the previous tile.
// ---------------------------------------------------------------------------
__global__ __launch_bounds__(256) void flash_attn_kernel(
    const _Float16* __restrict__ Qb, const _Float16* __restrict__ Kb,
    const _Float16* __restrict__ Vt, _Float16* __restrict__ Ob)
{
    __shared__ __align__(16) _Float16 Ks[2][64][64];   // [buf][key][dim]
    __shared__ __align__(16) _Float16 Vs[2][64][64];   // [buf][dim][key]
    __shared__ __align__(16) _Float16 Ps[8][16][64];   // per-wave P tile

    const int tid = threadIdx.x;
    const int wid = tid >> 5, lane = tid & 31;
    const int lanelo = lane & 15, halfsel = lane >> 4;
    const int y = blockIdx.y;                 // 0..31
    const int s = y >> 4, b = (y >> 3) & 1, h = y & 7;
    const size_t qhead = (size_t)((s * 2 + b) * 8 + h);
    const size_t khead = (size_t)(b * 8 + h); // stream-0 keys
    const int q0 = blockIdx.x * 128 + wid * 16;

    // Per-thread staging coordinates (16 halves of one row per buffer half).
    const int srow = tid >> 2;
    const int scc  = (tid & 3) * 16;
    const _Float16* kp0 = Kb + (khead * 4096 + srow) * 64 + scc;   // + kv*64
    const _Float16* vp0 = Vt + (qhead * 64 + srow) * 4096 + scc;   // + kv

    // Q A-fragments (16 queries x 64 dims = 2 frags), loaded once.
    v16h aq[2];
    {
        const _Float16* qp = Qb + (qhead * 4096 + q0 + lanelo) * 64;
        aq[0] = ldA(qp + halfsel * 8,      qp + 16 + halfsel * 8);
        aq[1] = ldA(qp + 32 + halfsel * 8, qp + 48 + halfsel * 8);
    }

    v8f o[4] = {};
    float m_run[8], l_run[8];
#pragma unroll
    for (int r = 0; r < 8; ++r) { m_run[r] = -1e30f; l_run[r] = 0.f; }

    // Kick off async staging of tile 0 into buffer 0.
    {
        unsigned lk = (unsigned)(__UINTPTR_TYPE__)&Ks[0][srow][scc];
        unsigned lv = (unsigned)(__UINTPTR_TYPE__)&Vs[0][srow][scc];
        async_ld_b128(lk,      kp0);
        async_ld_b128(lk + 16, kp0 + 8);
        async_ld_b128(lv,      vp0);
        async_ld_b128(lv + 16, vp0 + 8);
    }

    int cur = 0;
    for (int kv = 0; kv < 4096; kv += 64) {
        wait_async0();          // this wave's tile-kv writes landed in LDS
        __syncthreads();        // ... and every other wave's too

        if (kv + 64 < 4096) {   // prefetch next tile into the other buffer
            int nxt = cur ^ 1;
            unsigned lk = (unsigned)(__UINTPTR_TYPE__)&Ks[nxt][srow][scc];
            unsigned lv = (unsigned)(__UINTPTR_TYPE__)&Vs[nxt][srow][scc];
            const _Float16* kp = kp0 + (size_t)(kv + 64) * 64;
            const _Float16* vp = vp0 + (kv + 64);
            async_ld_b128(lk,      kp);
            async_ld_b128(lk + 16, kp + 8);
            async_ld_b128(lv,      vp);
            async_ld_b128(lv + 16, vp + 8);
        }

        // S = Q K^T : 4 key-chunks x (K=64 as 2x32)
        v8f s4[4];
#pragma unroll
        for (int t = 0; t < 4; ++t) {
            int key = t * 16 + lanelo;
            v16h b0 = ldA(&Ks[cur][key][halfsel * 8],      &Ks[cur][key][16 + halfsel * 8]);
            v16h b1 = ldA(&Ks[cur][key][32 + halfsel * 8], &Ks[cur][key][48 + halfsel * 8]);
            v8f z = {};
            z = wmma16(aq[0], b0, z);
            s4[t] = wmma16(aq[1], b1, z);
        }

        // Online softmax over this 64-key chunk (rows live across 16 lanes).
#pragma unroll
        for (int r = 0; r < 8; ++r) {
            float c = fmaxf(fmaxf(s4[0][r], s4[1][r]), fmaxf(s4[2][r], s4[3][r]));
            c = fmaxf(c, __shfl_xor(c, 1));
            c = fmaxf(c, __shfl_xor(c, 2));
            c = fmaxf(c, __shfl_xor(c, 4));
            c = fmaxf(c, __shfl_xor(c, 8));
            float mn = fmaxf(m_run[r], c);
            float alpha = __expf(m_run[r] - mn);
            m_run[r] = mn;
            l_run[r] *= alpha;
#pragma unroll
            for (int t = 0; t < 4; ++t) o[t][r] *= alpha;

            float rs = 0.f;
            int rowm = halfsel ? r + 8 : r;
#pragma unroll
            for (int t = 0; t < 4; ++t) {
                float p = __expf(s4[t][r] - mn);
                rs += p;
                Ps[wid][rowm][t * 16 + lanelo] = (_Float16)p;   // C-layout -> LDS
            }
            rs += __shfl_xor(rs, 1);
            rs += __shfl_xor(rs, 2);
            rs += __shfl_xor(rs, 4);
            rs += __shfl_xor(rs, 8);
            l_run[r] += rs;
        }

        // O += P V : reload P as A-fragments (16 x 64 keys = 2 frags).
        v16h ap0 = ldA(&Ps[wid][lanelo][halfsel * 8],      &Ps[wid][lanelo][16 + halfsel * 8]);
        v16h ap1 = ldA(&Ps[wid][lanelo][32 + halfsel * 8], &Ps[wid][lanelo][48 + halfsel * 8]);
#pragma unroll
        for (int t = 0; t < 4; ++t) {
            int dim = t * 16 + lanelo;
            v16h b0 = ldA(&Vs[cur][dim][halfsel * 8],      &Vs[cur][dim][16 + halfsel * 8]);
            v16h b1 = ldA(&Vs[cur][dim][32 + halfsel * 8], &Vs[cur][dim][48 + halfsel * 8]);
            o[t] = wmma16(ap0, b0, o[t]);
            o[t] = wmma16(ap1, b1, o[t]);
        }
        __syncthreads();        // readers done before this buffer is refilled
        cur ^= 1;
    }

    // Normalize and store O as f16 [s*2+b][n][h*64+d] for the proj GEMM.
#pragma unroll
    for (int r = 0; r < 8; ++r) {
        float inv = 1.0f / l_run[r];
        int n = q0 + (halfsel ? r + 8 : r);
        size_t rowoff = ((size_t)(s * 2 + b) * 4096 + n) * 512 + h * 64;
#pragma unroll
        for (int t = 0; t < 4; ++t)
            Ob[rowoff + t * 16 + lanelo] = (_Float16)(o[t][r] * inv);
    }
}

// ---------------------------------------------------------------------------
// Kernel 3: projection GEMM. O[16384 x 512] (f16) @ Wproj[512 x 512] + bias,
// fp32 directly into d_out (row order == (o1,o2) concatenation).
// ---------------------------------------------------------------------------
__global__ __launch_bounds__(256) void proj_gemm_kernel(
    const _Float16* __restrict__ A, const float* __restrict__ w,
    const float* __restrict__ bias, float* __restrict__ out)
{
    __shared__ __align__(16) _Float16 As[128][32];
    __shared__ __align__(16) _Float16 Bs[64][32];

    const int tid = threadIdx.x;
    const int wid = tid >> 5, lane = tid & 31;
    const int lanelo = lane & 15, halfsel = lane >> 4;
    const int wm = wid & 3, wn = wid >> 2;
    const int mb = blockIdx.x * 128;
    const int nb = blockIdx.y * 64;

    v8f acc[2][2] = {};

    for (int kb = 0; kb < 512; kb += 32) {
#pragma unroll
        for (int p = 0; p < 2; ++p) {
            int r = (tid >> 2) + p * 64;
            int c = (tid & 3) * 8;
            *(v8h*)&As[r][c] = *(const v8h*)(A + (size_t)(mb + r) * 512 + kb + c);
        }
#pragma unroll
        for (int p = 0; p < 8; ++p) {
            int n = tid & 63;
            int k = (tid >> 6) + p * 4;
            Bs[n][k] = (_Float16)w[(size_t)(kb + k) * 512 + nb + n];
        }
        __syncthreads();

        v16h a[2], bf[2];
#pragma unroll
        for (int i = 0; i < 2; ++i) {
            int row = wm * 32 + i * 16 + lanelo;
            a[i] = ldA(&As[row][halfsel * 8], &As[row][16 + halfsel * 8]);
        }
#pragma unroll
        for (int j = 0; j < 2; ++j) {
            int col = wn * 32 + j * 16 + lanelo;
            bf[j] = ldA(&Bs[col][halfsel * 8], &Bs[col][16 + halfsel * 8]);
        }
#pragma unroll
        for (int i = 0; i < 2; ++i)
#pragma unroll
            for (int j = 0; j < 2; ++j)
                acc[i][j] = wmma16(a[i], bf[j], acc[i][j]);
        __syncthreads();
    }

#pragma unroll
    for (int j = 0; j < 2; ++j) {
        int n = nb + wn * 32 + j * 16 + lanelo;
        float bj = bias[n];
#pragma unroll
        for (int i = 0; i < 2; ++i)
#pragma unroll
            for (int r = 0; r < 8; ++r) {
                int m = mb + wm * 32 + i * 16 + (halfsel ? r + 8 : r);
                out[(size_t)m * 512 + n] = acc[i][j][r] + bj;
            }
    }
}

// ---------------------------------------------------------------------------
extern "C" void kernel_launch(void* const* d_in, const int* in_sizes, int n_in,
                              void* d_out, int out_size, void* d_ws, size_t ws_size,
                              hipStream_t stream) {
    const float* x1    = (const float*)d_in[0];
    const float* x2    = (const float*)d_in[1];
    const float* wqkv  = (const float*)d_in[2];
    const float* wproj = (const float*)d_in[3];
    const float* bproj = (const float*)d_in[4];
    float* out = (float*)d_out;

    // Workspace partition: Q, K, V^T, O buffers, each 2*2*8*4096*64 f16
    // (16.78 MB) => 67.1 MB total (fits in the 192 MB L2).
    const size_t BUF = (size_t)2 * 2 * 8 * 4096 * 64;
    _Float16* Qb = (_Float16*)d_ws;
    _Float16* Kb = Qb + BUF;
    _Float16* Vt = Kb + BUF;
    _Float16* Ob = Vt + BUF;

    qkv_gemm_kernel<<<dim3(128, 24), 256, 0, stream>>>(x1, x2, wqkv, Qb, Kb, Vt);
    flash_attn_kernel<<<dim3(32, 32), 256, 0, stream>>>(Qb, Kb, Vt, Ob);
    proj_gemm_kernel<<<dim3(128, 8), 256, 0, stream>>>(Ob, wproj, bproj, out);
}
